// CausalSelfAttention_26225070310003
// MI455X (gfx1250) — matrix operand order repaired and checked by reading, not compiled, not run on hardware
//
#include <hip/hip_runtime.h>
#include <hip/hip_bf16.h>
#include <math.h>

// ---------------------------------------------------------------------------
// CDNA5 (gfx1250) wave32 WMMA types
// ---------------------------------------------------------------------------
typedef __attribute__((ext_vector_type(16))) __bf16    v16bf;
typedef __attribute__((ext_vector_type(8)))  float     v8f;
typedef __attribute__((ext_vector_type(4)))  unsigned  u32x4;
typedef __attribute__((ext_vector_type(8)))  unsigned  u32x8;

union AFrag { v16bf v; u32x4 q[2]; };
union BFrag { v16bf v; u32x4 q[2]; };

__device__ __forceinline__ v8f wmma_bf16(const AFrag& a, const BFrag& b, v8f c) {
    return __builtin_amdgcn_wmma_f32_16x16x32_bf16(
        false, a.v, false, b.v, (short)0, c, false, false);
}

// fp32 -> bf16, round-to-nearest-even
__device__ __forceinline__ __bf16 f2bf(float f) {
    unsigned u = __builtin_bit_cast(unsigned, f);
    unsigned r = (u + 0x7FFFu + ((u >> 16) & 1u)) >> 16;
    unsigned short h = (unsigned short)r;
    return __builtin_bit_cast(__bf16, h);
}

__device__ __forceinline__ unsigned pk2bf(float lo, float hi) {
    unsigned short l = __builtin_bit_cast(unsigned short, f2bf(lo));
    unsigned short h = __builtin_bit_cast(unsigned short, f2bf(hi));
    return (unsigned)l | ((unsigned)h << 16);
}

// ---------------------------------------------------------------------------
// Problem constants
// ---------------------------------------------------------------------------
#define BSZ     4
#define SEQ     2048
#define DMODEL  1024
#define NHEAD   16
#define HDIM    64
#define MTOT    (BSZ * SEQ)          // 8192 rows
#define KCHUNK  128                  // K elements staged per TDM panel
#define NCHUNK  (DMODEL / KCHUNK)    // 8
#define LDB     136                  // padded LDS row stride: 128 + 8 elems (272 B)
#define PTLD    40                   // padded P-tile row stride: 32 + 8 elems (80 B)

// ---------------------------------------------------------------------------
// fp32 -> bf16 conversion (grid-stride)
// ---------------------------------------------------------------------------
__global__ void cvt_f32_bf16(const float* __restrict__ in,
                             __bf16* __restrict__ out, int n) {
    int i = blockIdx.x * blockDim.x + threadIdx.x;
    int stride = gridDim.x * blockDim.x;
    for (; i < n; i += stride) out[i] = f2bf(in[i]);
}

// ---------------------------------------------------------------------------
// TDM: DMA one B panel (64 rows x 128 K bf16, global row stride 1024 elems)
// into LDS with hardware padding: 4 DWORDs (16 B) inserted every 64 DWORDs
// (one 256 B row) -> LDS row stride 272 B = 68 DWORDs -> conflict-free
// ds_load_b128 (lane n starts at bank 4n mod 64).
// D# group0: count=1 | lds_addr | global_addr[56:0] | type=2
// D# group1: data_size=2B | pad_en, pad_interval=5(64 DW), pad_amount=3(4 DW)
//            tensor 128x64, tile 128x64, dim0_stride=1024
// ---------------------------------------------------------------------------
__device__ __forceinline__ void tdm_load_panel(const __bf16* gsrc, __bf16* ldst) {
    unsigned long long ga = (unsigned long long)(uintptr_t)gsrc;
    unsigned lds = (unsigned)(uintptr_t)ldst;
    u32x4 g0 = { 1u,                                       // count=1, user mode
                 lds,                                      // lds_addr
                 (unsigned)ga,                             // global_addr[31:0]
                 (unsigned)((ga >> 32) & 0x01FFFFFFull)    // global_addr[56:32]
                 | 0x80000000u };                          // type=2 ("image")
    u32x8 g1 = { 0x00010000u                               // data_size=1 (2 bytes)
                 | (1u << 20)                              // pad_enable
                 | (5u << 22)                              // pad_interval: 64 DWORDs
                 | (3u << 25),                             // pad_amount: 4 DWORDs
                 (128u << 16),       // tensor_dim0[15:0]=128 at bits[63:48]
                 (64u  << 16),       // tensor_dim1[15:0]=64 at bits[95:80]
                 (128u << 16),       // tile_dim0=128 at bits[127:112]
                 64u,                // tile_dim1=64, tile_dim2=0
                 1024u,              // tensor_dim0_stride[31:0] = 1024 elems
                 0u, 0u };
    asm volatile("tensor_load_to_lds %0, %1" :: "s"(g0), "s"(g1) : "memory");
}

// ---------------------------------------------------------------------------
// Block GEMM core: 8 waves/block, each wave 32(M)x64(N); waves share the
// 64-row B panel staged in double-buffered LDS by the TDM.
//   A: row-major [M,K=1024] bf16.  W: row-major [N,K=1024] bf16 (we do A*W^T).
// acc[0..3]: rows m0w..m0w+15, subtiles n0+16*sub ; acc[4..7]: rows +16.
// ---------------------------------------------------------------------------
__device__ __forceinline__ void gemm32x64_tdm(const __bf16* __restrict__ A0,
                                              const __bf16* __restrict__ Wn0,
                                              __bf16* lbuf0, __bf16* lbuf1,
                                              int wslot, int lane, v8f acc[8]) {
    const int am    = lane & 15;
    const int akoff = (lane >> 4) << 4;   // 0 / 16
    const int bn    = lane & 15;
    const int bkoff = (lane >> 4) << 4;   // 0 / 16
    const __bf16* ar0 = A0 + (size_t)am * DMODEL + akoff;
    const __bf16* ar1 = ar0 + 16 * DMODEL;
    __bf16* lbuf[2] = { lbuf0, lbuf1 };

    if (wslot == 0) tdm_load_panel(Wn0, lbuf[0]);

    for (int c = 0; c < NCHUNK; ++c) {
        if (wslot == 0) {
            if (c + 1 < NCHUNK) {
                tdm_load_panel(Wn0 + (c + 1) * KCHUNK, lbuf[(c + 1) & 1]);
                __builtin_amdgcn_s_wait_tensorcnt(1);   // chunk c landed
            } else {
                __builtin_amdgcn_s_wait_tensorcnt(0);
            }
        }
        __syncthreads();                                // panel c visible to all
        const __bf16* bp  = lbuf[c & 1];
        const __bf16* ak0 = ar0 + c * KCHUNK;
        const __bf16* ak1 = ar1 + c * KCHUNK;
#pragma unroll
        for (int kk = 0; kk < KCHUNK; kk += 32) {
            AFrag a0, a1;
            a0.q[0] = *(const u32x4*)(ak0 + kk);
            a0.q[1] = *(const u32x4*)(ak0 + kk + 8);
            a1.q[0] = *(const u32x4*)(ak1 + kk);
            a1.q[1] = *(const u32x4*)(ak1 + kk + 8);
#pragma unroll
            for (int sub = 0; sub < 4; ++sub) {
                BFrag b;
                const __bf16* p = bp + (size_t)(sub * 16 + bn) * LDB + kk + bkoff;
                b.q[0] = *(const u32x4*)(p);
                b.q[1] = *(const u32x4*)(p + 8);
                acc[sub]     = wmma_bf16(a0, b, acc[sub]);
                acc[4 + sub] = wmma_bf16(a1, b, acc[4 + sub]);
            }
        }
        __syncthreads();                                // done with panel c&1
    }
}

// ---------------------------------------------------------------------------
// GEMM 1: qkv = x @ w_qkv^T, scattered into head-major Q, K and transposed V
//   qh,kh: [bh][s][hd]   (bh = b*16+h)      vt: [bh][hd][s]
// Block: 256(M) x 64(N); the 64-wide N strip == one head's 64 dims.
// ---------------------------------------------------------------------------
__global__ __launch_bounds__(256) void gemm_qkv(
    const __bf16* __restrict__ X, const __bf16* __restrict__ W,
    __bf16* __restrict__ qh, __bf16* __restrict__ kh, __bf16* __restrict__ vt) {
    __shared__ __bf16 bpanel[2][64 * LDB];      // 2 x 17 KB (padded rows)
    const int TNB = (3 * DMODEL) / 64;          // 48 N strips
    int wslot = threadIdx.x >> 5;
    int lane  = threadIdx.x & 31;
    int nIdx  = blockIdx.x % TNB;
    int mIdx  = blockIdx.x / TNB;
    int n0    = nIdx << 6;
    int m0w   = (mIdx << 8) + (wslot << 5);     // this wave's 32 rows

    v8f acc[8] = {v8f{}, v8f{}, v8f{}, v8f{}, v8f{}, v8f{}, v8f{}, v8f{}};
    gemm32x64_tdm(X + (size_t)m0w * DMODEL, W + (size_t)n0 * DMODEL,
                  &bpanel[0][0], &bpanel[1][0], wslot, lane, acc);

    int b     = m0w >> 11;                 // row / SEQ
    int sect  = n0 >> 10;                  // 0=q 1=k 2=v
    int h     = (n0 & (DMODEL - 1)) >> 6;
    int bh    = b * NHEAD + h;
    int col   = lane & 15;
    int rbase = (lane >> 4) << 3;          // rows r / r+8

#pragma unroll
    for (int hb = 0; hb < 2; ++hb) {       // two 16-row half tiles
        int s0 = (m0w + hb * 16) & (SEQ - 1);
        if (sect < 2) {
            __bf16* dst = (sect == 0) ? qh : kh;
#pragma unroll
            for (int sub = 0; sub < 4; ++sub) {
                v8f a = acc[hb * 4 + sub];
                int hd = sub * 16 + col;
#pragma unroll
                for (int r = 0; r < 8; ++r) {
                    int s = s0 + rbase + r;
                    dst[((size_t)bh * SEQ + s) * HDIM + hd] = f2bf(a[r]);
                }
            }
        } else {
            // transposed store: lane holds fixed hd, 8 consecutive s -> one b128
#pragma unroll
            for (int sub = 0; sub < 4; ++sub) {
                v8f a = acc[hb * 4 + sub];
                int hd = sub * 16 + col;
                u32x4 pk;
                pk.x = pk2bf(a[0], a[1]);
                pk.y = pk2bf(a[2], a[3]);
                pk.z = pk2bf(a[4], a[5]);
                pk.w = pk2bf(a[6], a[7]);
                *(u32x4*)(vt + ((size_t)bh * HDIM + hd) * SEQ + s0 + rbase) = pk;
            }
        }
    }
}

// ---------------------------------------------------------------------------
// Flash-attention per wave: one 16-query tile, stream 32-key blocks (causal).
// ---------------------------------------------------------------------------
__global__ __launch_bounds__(256) void attn_fwd(
    const __bf16* __restrict__ qh, const __bf16* __restrict__ kh,
    const __bf16* __restrict__ vt, __bf16* __restrict__ ab) {
    __shared__ __bf16 ptile[8][16 * PTLD];  // per-wave P restage tile (padded)

    int wslot = threadIdx.x >> 5;
    int lane  = threadIdx.x & 31;
    int wid   = blockIdx.x * 8 + wslot;
    int qb    = wid & (SEQ / 16 - 1);      // 128 q-tiles per head
    int bh    = wid >> 7;
    int b     = bh >> 4, h = bh & 15;

    const int am    = lane & 15;
    const int akoff = (lane >> 4) << 4;
    const int bn    = lane & 15;
    const int bkoff = (lane >> 4) << 4;
    const int rbase = (lane >> 4) << 3;

    // Q fragments: 16x64 = two 16x32 A tiles (resident all loop)
    const __bf16* qbase = qh + ((size_t)bh * SEQ + qb * 16) * HDIM;
    AFrag qa0, qa1;
    qa0.q[0] = *(const u32x4*)(qbase + (size_t)am * HDIM + akoff);
    qa0.q[1] = *(const u32x4*)(qbase + (size_t)am * HDIM + akoff + 8);
    qa1.q[0] = *(const u32x4*)(qbase + (size_t)am * HDIM + 32 + akoff);
    qa1.q[1] = *(const u32x4*)(qbase + (size_t)am * HDIM + 32 + akoff + 8);

    v8f o0 = {}, o1 = {}, o2 = {}, o3 = {};
    float mrow[8], lrow[8];
#pragma unroll
    for (int r = 0; r < 8; ++r) { mrow[r] = -1e30f; lrow[r] = 0.f; }

    const __bf16* kbase = kh + (size_t)bh * SEQ * HDIM;
    const __bf16* vbase = vt + (size_t)bh * HDIM * SEQ;
    __bf16* pt = &ptile[wslot][0];

    const int nkb = (qb * 16 + 47) >> 5;   // key blocks up to causal frontier
    for (int kb = 0; kb < nkb; ++kb) {
        int k0 = kb << 5;

        // ---- S = Q K^T: two 16-key subtiles, K=64 in 2 steps
        v8f s0 = {}, s1 = {};
        {
            const __bf16* kr = kbase + (size_t)(k0 + bn) * HDIM + bkoff;
            BFrag f0, f1;
            f0.q[0] = *(const u32x4*)(kr);
            f0.q[1] = *(const u32x4*)(kr + 8);
            f1.q[0] = *(const u32x4*)(kr + 32);
            f1.q[1] = *(const u32x4*)(kr + 40);
            s0 = wmma_bf16(qa0, f0, s0);
            s0 = wmma_bf16(qa1, f1, s0);
            const __bf16* kr2 = kr + 16 * HDIM;
            BFrag g0, g1;
            g0.q[0] = *(const u32x4*)(kr2);
            g0.q[1] = *(const u32x4*)(kr2 + 8);
            g1.q[0] = *(const u32x4*)(kr2 + 32);
            g1.q[1] = *(const u32x4*)(kr2 + 40);
            s1 = wmma_bf16(qa0, g0, s1);
            s1 = wmma_bf16(qa1, g1, s1);
        }

        // ---- online softmax (rows live across 16-lane groups)
#pragma unroll
        for (int r = 0; r < 8; ++r) {
            int srow = qb * 16 + rbase + r;
            int key0 = k0 + (lane & 15);
            float e0 = s0[r] * 0.125f;                 // 1/sqrt(64)
            float e1 = s1[r] * 0.125f;
            if (key0 > srow)      e0 = -1e30f;         // causal mask
            if (key0 + 16 > srow) e1 = -1e30f;
            float mx = fmaxf(e0, e1);
            mx = fmaxf(mx, __shfl_xor(mx, 1));
            mx = fmaxf(mx, __shfl_xor(mx, 2));
            mx = fmaxf(mx, __shfl_xor(mx, 4));
            mx = fmaxf(mx, __shfl_xor(mx, 8));
            float mnew = fmaxf(mrow[r], mx);
            float corr = __expf(mrow[r] - mnew);
            mrow[r] = mnew;
            float p0 = __expf(e0 - mnew);
            float p1 = __expf(e1 - mnew);
            float rs = p0 + p1;
            rs += __shfl_xor(rs, 1);
            rs += __shfl_xor(rs, 2);
            rs += __shfl_xor(rs, 4);
            rs += __shfl_xor(rs, 8);
            lrow[r] = lrow[r] * corr + rs;
            o0[r] *= corr; o1[r] *= corr; o2[r] *= corr; o3[r] *= corr;
            int rr = rbase + r;
            pt[rr * PTLD + (lane & 15)]      = f2bf(p0);
            pt[rr * PTLD + 16 + (lane & 15)] = f2bf(p1);
        }

        // same-wave LDS RAW: DS ops are in-order; fence compiler + counter
        asm volatile("s_wait_dscnt 0" ::: "memory");

        // ---- reload P in A-layout (16x32 over keys)
        AFrag pa;
        pa.q[0] = *(const u32x4*)(pt + am * PTLD + akoff);
        pa.q[1] = *(const u32x4*)(pt + am * PTLD + akoff + 8);

        // ---- O += P * V  (V pre-transposed: [hd][s], contiguous keys)
        {
            const __bf16* vr0 = vbase + (size_t)(0 * 16 + bn) * SEQ + k0 + bkoff;
            const __bf16* vr1 = vbase + (size_t)(1 * 16 + bn) * SEQ + k0 + bkoff;
            const __bf16* vr2 = vbase + (size_t)(2 * 16 + bn) * SEQ + k0 + bkoff;
            const __bf16* vr3 = vbase + (size_t)(3 * 16 + bn) * SEQ + k0 + bkoff;
            BFrag v0, v1, v2, v3;
            v0.q[0] = *(const u32x4*)(vr0); v0.q[1] = *(const u32x4*)(vr0 + 8);
            v1.q[0] = *(const u32x4*)(vr1); v1.q[1] = *(const u32x4*)(vr1 + 8);
            v2.q[0] = *(const u32x4*)(vr2); v2.q[1] = *(const u32x4*)(vr2 + 8);
            v3.q[0] = *(const u32x4*)(vr3); v3.q[1] = *(const u32x4*)(vr3 + 8);
            o0 = wmma_bf16(pa, v0, o0);
            o1 = wmma_bf16(pa, v1, o1);
            o2 = wmma_bf16(pa, v2, o2);
            o3 = wmma_bf16(pa, v3, o3);
        }
    }

    // ---- normalize and store into [b][s][d_model] bf16 for the proj GEMM
    size_t obase = ((size_t)b * SEQ + qb * 16) * DMODEL + h * HDIM;
#pragma unroll
    for (int r = 0; r < 8; ++r) {
        float inv = 1.0f / lrow[r];
        int srow = rbase + r;
        __bf16* orow = ab + obase + (size_t)srow * DMODEL + (lane & 15);
        orow[0]  = f2bf(o0[r] * inv);
        orow[16] = f2bf(o1[r] * inv);
        orow[32] = f2bf(o2[r] * inv);
        orow[48] = f2bf(o3[r] * inv);
    }
}

// ---------------------------------------------------------------------------
// GEMM 2: out = attn_out @ w_proj^T, fp32 output
// ---------------------------------------------------------------------------
__global__ __launch_bounds__(256) void gemm_proj(
    const __bf16* __restrict__ A, const __bf16* __restrict__ W,
    float* __restrict__ out) {
    __shared__ __bf16 bpanel[2][64 * LDB];      // 2 x 17 KB (padded rows)
    const int TNB = DMODEL / 64;                // 16 N strips
    int wslot = threadIdx.x >> 5;
    int lane  = threadIdx.x & 31;
    int nIdx  = blockIdx.x % TNB;
    int mIdx  = blockIdx.x / TNB;
    int n0    = nIdx << 6;
    int m0w   = (mIdx << 8) + (wslot << 5);

    v8f acc[8] = {v8f{}, v8f{}, v8f{}, v8f{}, v8f{}, v8f{}, v8f{}, v8f{}};
    gemm32x64_tdm(A + (size_t)m0w * DMODEL, W + (size_t)n0 * DMODEL,
                  &bpanel[0][0], &bpanel[1][0], wslot, lane, acc);

    int col   = lane & 15;
    int rbase = (lane >> 4) << 3;
#pragma unroll
    for (int hb = 0; hb < 2; ++hb) {
        int m0 = m0w + hb * 16;
#pragma unroll
        for (int sub = 0; sub < 4; ++sub) {
            v8f a = acc[hb * 4 + sub];
#pragma unroll
            for (int r = 0; r < 8; ++r) {
                out[(size_t)(m0 + rbase + r) * DMODEL + n0 + sub * 16 + col] = a[r];
            }
        }
    }
}

// ---------------------------------------------------------------------------
// Host entry
// ---------------------------------------------------------------------------
extern "C" void kernel_launch(void* const* d_in, const int* in_sizes, int n_in,
                              void* d_out, int out_size, void* d_ws, size_t ws_size,
                              hipStream_t stream) {
    (void)in_sizes; (void)n_in; (void)out_size; (void)ws_size;
    const float* x     = (const float*)d_in[0];   // [4,2048,1024]
    const float* wqkv  = (const float*)d_in[1];   // [3072,1024]
    const float* wproj = (const float*)d_in[2];   // [1024,1024]
    float* out = (float*)d_out;                   // [4,2048,1024] fp32

    char* ws = (char*)d_ws;
    size_t off = 0;
    auto alloc = [&](size_t bytes) -> void* {
        void* p = ws + off;
        off += (bytes + 255) & ~(size_t)255;
        return p;
    };
    __bf16* xb     = (__bf16*)alloc((size_t)MTOT * DMODEL * 2);        // 16 MB
    __bf16* wqkvb  = (__bf16*)alloc((size_t)3 * DMODEL * DMODEL * 2);  // 6 MB
    __bf16* wprojb = (__bf16*)alloc((size_t)DMODEL * DMODEL * 2);      // 2 MB
    __bf16* q      = (__bf16*)alloc((size_t)BSZ * NHEAD * SEQ * HDIM * 2);
    __bf16* k      = (__bf16*)alloc((size_t)BSZ * NHEAD * SEQ * HDIM * 2);
    __bf16* vt     = (__bf16*)alloc((size_t)BSZ * NHEAD * SEQ * HDIM * 2);
    __bf16* ab     = (__bf16*)alloc((size_t)MTOT * DMODEL * 2);        // 16 MB

    int nx = MTOT * DMODEL, nq = 3 * DMODEL * DMODEL, np = DMODEL * DMODEL;
    cvt_f32_bf16<<<4096, 256, 0, stream>>>(x, xb, nx);
    cvt_f32_bf16<<<4096, 256, 0, stream>>>(wqkv, wqkvb, nq);
    cvt_f32_bf16<<<2048, 256, 0, stream>>>(wproj, wprojb, np);

    // 32 M-blocks(256 rows) * 48 N-strips
    gemm_qkv<<<1536, 256, 0, stream>>>(xb, wqkvb, q, k, vt);
    // 64 heads * 128 q-tiles = 8192 waves, 8 waves/block
    attn_fwd<<<1024, 256, 0, stream>>>(q, k, vt, ab);
    // 32 M-blocks * 16 N-strips
    gemm_proj<<<512, 256, 0, stream>>>(ab, wprojb, out);
}
